// SurfaceTextureInpaintingNet_13795434955522
// MI455X (gfx1250) — compile-verified, matrix operand order checked by reading, not run on hardware
//
#include <hip/hip_runtime.h>
#include <math.h>

// ---------------- problem constants (match reference) ----------------
#define N0C 80000
#define N1C 15000
#define N2C 3750
#define E0C 1280000
#define E1C 240000
#define E2C 60000
#define NGF 64

typedef __attribute__((ext_vector_type(16))) _Float16 v16h;
typedef __attribute__((ext_vector_type(8)))  _Float16 v8h;
typedef __attribute__((ext_vector_type(8)))  float    v8f;

// ---------------- elementwise / scatter kernels ----------------

__global__ void k_zero(float* __restrict__ p, long long count) {
    long long i = (long long)blockIdx.x * blockDim.x + threadIdx.x;
    long long stride = (long long)gridDim.x * blockDim.x;
    for (; i < count; i += stride) p[i] = 0.0f;
}

__global__ void k_count(const int* __restrict__ idx, int n, float* __restrict__ cnt) {
    int i = blockIdx.x * blockDim.x + threadIdx.x;
    if (i < n) atomicAdd(&cnt[idx[i]], 1.0f);
}

// one block per edge, one thread per channel: agg[dst[e], c] += x[src[e], c]
__global__ void k_edge_scatter(const float* __restrict__ x, const int* __restrict__ src,
                               const int* __restrict__ dst, float* __restrict__ agg) {
    int e = blockIdx.x;
    int c = threadIdx.x;
    int C = blockDim.x;
    atomicAdd(&agg[(size_t)dst[e] * C + c], x[(size_t)src[e] * C + c]);
}

// one block per row: out[seg[i], c] += x[i, c]
__global__ void k_seg_scatter(const float* __restrict__ x, const int* __restrict__ seg,
                              float* __restrict__ out) {
    int i = blockIdx.x;
    int c = threadIdx.x;
    int C = blockDim.x;
    atomicAdd(&out[(size_t)seg[i] * C + c], x[(size_t)i * C + c]);
}

// one block per row: agg[i, c] /= max(cnt[i], 1)
__global__ void k_div_cnt(float* __restrict__ agg, const float* __restrict__ cnt) {
    int i = blockIdx.x;
    int c = threadIdx.x;
    int C = blockDim.x;
    agg[(size_t)i * C + c] /= fmaxf(cnt[i], 1.0f);
}

// one block per row: out[i,c] = coarse[tr[i], c] + skip[i, c]
__global__ void k_gather_add(const float* __restrict__ coarse, const int* __restrict__ tr,
                             const float* __restrict__ skip, float* __restrict__ out) {
    int i = blockIdx.x;
    int c = threadIdx.x;
    int C = blockDim.x;
    out[(size_t)i * C + c] = coarse[(size_t)tr[i] * C + c] + skip[(size_t)i * C + c];
}

// f32 -> f16 conversion (activation staging for WMMA)
__global__ void k_cvt_f16(const float* __restrict__ in, _Float16* __restrict__ out,
                          long long nelem) {
    long long t = (long long)blockIdx.x * blockDim.x + threadIdx.x;
    if (t < nelem) out[t] = (_Float16)in[t];
}

// per-channel sum / sumsq partials. blockDim.x == C (64/128/256).
__global__ void k_stats(const float* __restrict__ x, int n, int rows_per_block,
                        float* __restrict__ stats) {
    int C = blockDim.x;
    int c = threadIdx.x;
    int r0 = blockIdx.x * rows_per_block;
    int r1 = r0 + rows_per_block;
    if (r1 > n) r1 = n;
    float s = 0.0f, s2 = 0.0f;
    for (int r = r0; r < r1; ++r) {
        float v = x[(size_t)r * C + c];
        s += v; s2 += v * v;
    }
    atomicAdd(&stats[c], s);
    atomicAdd(&stats[C + c], s2);
}

// one block per row: out[i,c] = res[i,c] + elu((T[i,c]-mean_c)*rsqrt(var_c+eps))
__global__ void k_norm_elu_res(const float* __restrict__ T, const float* __restrict__ res,
                               const float* __restrict__ stats, int n,
                               float* __restrict__ out) {
    int i = blockIdx.x;
    int c = threadIdx.x;
    int C = blockDim.x;
    size_t t = (size_t)i * C + c;
    float inv_n = 1.0f / (float)n;
    float mean = stats[c] * inv_n;
    float var  = stats[C + c] * inv_n - mean * mean;
    float r    = rsqrtf(var + 1e-5f);
    float v    = (T[t] - mean) * r;
    float e    = (v > 0.0f) ? v : expm1f(v);
    out[t] = (res ? res[t] : 0.0f) + e;
}

__global__ void k_tanh(float* __restrict__ p, long long n) {
    long long t = (long long)blockIdx.x * blockDim.x + threadIdx.x;
    if (t < n) p[t] = tanhf(p[t]);
}

// ---------------- weight packing for WMMA ----------------
// Pack W (din x dout, f32, row-major) into fragment-ordered f16 so each lane of
// a wave loads its whole 32-byte B fragment contiguously:
//   Wp[((tn*KT + kt)*32 + lane)*16 + i] = f16(W[k, col])
//   k = kt*32 + (lane>>4)*8 + (i&7) + (i>=8 ? 16 : 0), col = tn*16 + (lane&15)
__global__ void k_packW(const float* __restrict__ W, _Float16* __restrict__ Wp,
                        int din, int dout, int KT, int NT) {
    int t = blockIdx.x * blockDim.x + threadIdx.x;
    int total = NT * KT * 512;
    if (t >= total) return;
    int i    = t & 15;
    int lane = (t >> 4) & 31;
    int frag = t >> 9;          // tn*KT + kt
    int kt   = frag % KT;
    int tn   = frag / KT;
    int hi = lane >> 4, lm = lane & 15;
    int ka  = kt * 32 + hi * 8 + (i & 7) + ((i >> 3) << 4);
    int col = tn * 16 + lm;
    float v = (ka < din && col < dout) ? W[(size_t)ka * dout + col] : 0.0f;
    Wp[t] = (_Float16)v;
}

// ---------------- WMMA GEMM (din % 32 == 0) ----------------
// Y[n x dout] = (acc ? Y : 0) + Xh[n x din] @ W + (bias ? bias : 0)
// Xh: f16 row-major activations; Wp: fragment-packed f16 weights.
// One wave computes a 64x16 strip: 4 stacked 16x16 WMMA tiles sharing one B
// fragment per K-step. All fragment loads for a K-step are issued before the
// four WMMAs so the compiler can use partial loadcnt waits and overlap load
// latency with matrix-pipe execution. Rows are clamped; only stores are masked.
__global__ __launch_bounds__(256)
void k_gemm(const _Float16* __restrict__ Xh, const _Float16* __restrict__ Wp,
            const float* __restrict__ bias, float* __restrict__ Y,
            int n, int din, int dout, int acc) {
    int wave = (int)((blockIdx.x * blockDim.x + threadIdx.x) >> 5);
    int lane = threadIdx.x & 31;
    int lm = lane & 15;
    int hi = lane >> 4;
    int KT = din >> 5;
    int NT = (dout + 15) >> 4;
    int tiles_m = (n + 63) >> 6;
    int tm = wave / NT;
    int tn = wave - tm * NT;
    if (tm >= tiles_m) return;            // whole-wave uniform exit
    int row_base = tm << 6;
    int ncol  = (tn << 4) + lm;
    int ncolc = (ncol < dout) ? ncol : (dout - 1);

    // per-subtile A base pointers (row clamped into range), advanced by K-step
    const _Float16* ap0;
    const _Float16* ap1;
    const _Float16* ap2;
    const _Float16* ap3;
    {
        int r0 = row_base + 0 * 16 + lm; if (r0 > n - 1) r0 = n - 1;
        int r1 = row_base + 1 * 16 + lm; if (r1 > n - 1) r1 = n - 1;
        int r2 = row_base + 2 * 16 + lm; if (r2 > n - 1) r2 = n - 1;
        int r3 = row_base + 3 * 16 + lm; if (r3 > n - 1) r3 = n - 1;
        unsigned kb = (unsigned)(hi * 8);
        ap0 = Xh + (unsigned)r0 * (unsigned)din + kb;
        ap1 = Xh + (unsigned)r1 * (unsigned)din + kb;
        ap2 = Xh + (unsigned)r2 * (unsigned)din + kb;
        ap3 = Xh + (unsigned)r3 * (unsigned)din + kb;
    }
    const _Float16* bp = Wp + ((size_t)tn * KT * 32 + lane) * 16;

    v8f c0 = v8f{0.f,0.f,0.f,0.f,0.f,0.f,0.f,0.f};
    v8f c1 = c0, c2 = c0, c3 = c0;

    for (int kt = 0; kt < KT; ++kt) {
        // issue every load for this K-step first (loads complete in order)
        v16h b  = *(const v16h*)bp;
        v8h a00 = *(const v8h*)(ap0);
        v8h a01 = *(const v8h*)(ap0 + 16);
        v8h a10 = *(const v8h*)(ap1);
        v8h a11 = *(const v8h*)(ap1 + 16);
        v8h a20 = *(const v8h*)(ap2);
        v8h a21 = *(const v8h*)(ap2 + 16);
        v8h a30 = *(const v8h*)(ap3);
        v8h a31 = *(const v8h*)(ap3 + 16);
        bp += 512;
        ap0 += 32; ap1 += 32; ap2 += 32; ap3 += 32;

        v16h a;
        a = __builtin_shufflevector(a00, a01, 0,1,2,3,4,5,6,7, 8,9,10,11,12,13,14,15);
        c0 = __builtin_amdgcn_wmma_f32_16x16x32_f16(false, a, false, b, (short)0, c0, false, false);
        a = __builtin_shufflevector(a10, a11, 0,1,2,3,4,5,6,7, 8,9,10,11,12,13,14,15);
        c1 = __builtin_amdgcn_wmma_f32_16x16x32_f16(false, a, false, b, (short)0, c1, false, false);
        a = __builtin_shufflevector(a20, a21, 0,1,2,3,4,5,6,7, 8,9,10,11,12,13,14,15);
        c2 = __builtin_amdgcn_wmma_f32_16x16x32_f16(false, a, false, b, (short)0, c2, false, false);
        a = __builtin_shufflevector(a30, a31, 0,1,2,3,4,5,6,7, 8,9,10,11,12,13,14,15);
        c3 = __builtin_amdgcn_wmma_f32_16x16x32_f16(false, a, false, b, (short)0, c3, false, false);
    }

    // epilogue: C/D layout VGPR r -> M=r (lanes 0-15) / M=8+r (lanes 16-31)
    float bv = bias ? bias[ncolc] : 0.0f;
    v8f cc[4] = { c0, c1, c2, c3 };
    #pragma unroll
    for (int s = 0; s < 4; ++s) {
        #pragma unroll
        for (int r8 = 0; r8 < 8; ++r8) {
            int outrow = row_base + s * 16 + r8 + hi * 8;
            if (outrow < n && ncol < dout) {
                float v = cc[s][r8] + bv;
                size_t idx = (size_t)outrow * dout + ncol;
                if (acc) v += Y[idx];
                Y[idx] = v;
            }
        }
    }
}

// ---------------- small-K GEMM (din < 32, e.g. din = 3): plain FMA ----------------
__global__ void k_gemm_small(const float* __restrict__ X, const float* __restrict__ W,
                             const float* __restrict__ bias, float* __restrict__ Y,
                             int n, int din, int dout, int acc) {
    long long t = (long long)blockIdx.x * blockDim.x + threadIdx.x;
    if (t >= (long long)n * dout) return;
    int i = (int)(t / dout);
    int j = (int)(t - (long long)i * dout);
    float s = bias ? bias[j] : 0.0f;
    for (int k = 0; k < din; ++k)
        s += X[(size_t)i * din + k] * W[(size_t)k * dout + j];
    if (acc) s += Y[t];
    Y[t] = s;
}

// ---------------- host-side orchestration helpers ----------------

static inline unsigned blocksFor(long long total, int bs) {
    return (unsigned)((total + bs - 1) / bs);
}

static void gemm16(const _Float16* Xh, const float* W, const float* bias, float* Y,
                   int n, int din, int dout, int acc, _Float16* wpack, hipStream_t s) {
    int KT = din >> 5;
    int NT = (dout + 15) >> 4;
    int total = NT * KT * 512;
    k_packW<<<blocksFor(total, 256), 256, 0, s>>>(W, wpack, din, dout, KT, NT);
    long long waves = (long long)((n + 63) >> 6) * NT;
    k_gemm<<<blocksFor(waves * 32, 256), 256, 0, s>>>(Xh, wpack, bias, Y, n, din, dout, acc);
}

static void gemm_small(const float* X, const float* W, const float* bias, float* Y,
                       int n, int din, int dout, int acc, hipStream_t s) {
    k_gemm_small<<<blocksFor((long long)n * dout, 256), 256, 0, s>>>(
        X, W, bias, Y, n, din, dout, acc);
}

// SAGE conv + instance-norm + ELU + residual (optionally shortcut-projected).
static void run_block(const float* x, const int* edges, long long E, const float* cnt,
                      const float* Wl, const float* bl, const float* Wr,
                      const float* Ws, const float* bs,
                      int n, int din, int dout,
                      float* agg, float* T, float* Sbuf, float* stats, float* out,
                      _Float16* Ha, _Float16* Hb, _Float16* wpack, hipStream_t s) {
    long long aggN = (long long)n * din;
    k_zero<<<blocksFor(aggN, 256), 256, 0, s>>>(agg, aggN);
    k_edge_scatter<<<(unsigned)E, din, 0, s>>>(x, edges, edges + E, agg);
    k_div_cnt<<<(unsigned)n, din, 0, s>>>(agg, cnt);

    bool wide = (din & 31) == 0;
    if (wide) {
        k_cvt_f16<<<blocksFor(aggN, 256), 256, 0, s>>>(agg, Hb, aggN);
        k_cvt_f16<<<blocksFor(aggN, 256), 256, 0, s>>>(x,   Ha, aggN);
        gemm16(Hb, Wl, bl,      T, n, din, dout, 0, wpack, s);  // T  = agg @ Wl + bl
        gemm16(Ha, Wr, nullptr, T, n, din, dout, 1, wpack, s);  // T += x @ Wr
    } else {
        gemm_small(agg, Wl, bl,      T, n, din, dout, 0, s);
        gemm_small(x,   Wr, nullptr, T, n, din, dout, 1, s);
    }

    k_zero<<<1, 256, 0, s>>>(stats, 2 * dout);
    k_stats<<<blocksFor(n, 512), dout, 0, s>>>(T, n, 512, stats);

    const float* res = x;
    if (Ws) {
        if (wide) gemm16(Ha, Ws, bs, Sbuf, n, din, dout, 0, wpack, s);
        else      gemm_small(x, Ws, bs, Sbuf, n, din, dout, 0, s);
        res = Sbuf;
    }
    k_norm_elu_res<<<(unsigned)n, dout, 0, s>>>(T, res, stats, n, out);
}

static void seg_mean(const float* x, const int* seg, const float* tcnt,
                     int nin, int nout, int C, float* out, hipStream_t s) {
    long long outN = (long long)nout * C;
    k_zero<<<blocksFor(outN, 256), 256, 0, s>>>(out, outN);
    k_seg_scatter<<<(unsigned)nin, C, 0, s>>>(x, seg, out);
    k_div_cnt<<<(unsigned)nout, C, 0, s>>>(out, tcnt);
}

// ---------------- entry point ----------------

extern "C" void kernel_launch(void* const* d_in, const int* in_sizes, int n_in,
                              void* d_out, int out_size, void* d_ws, size_t ws_size,
                              hipStream_t stream) {
    (void)in_sizes; (void)n_in; (void)out_size; (void)ws_size;

    const float* x      = (const float*)d_in[0];
    const int*   edges0 = (const int*)d_in[1];
    const int*   edges1 = (const int*)d_in[2];
    const int*   edges2 = (const int*)d_in[3];
    const int*   trace0 = (const int*)d_in[4];
    const int*   trace1 = (const int*)d_in[5];
    const float* P[41];
    for (int i = 6; i < 41; ++i) P[i] = (const float*)d_in[i];
    // 6..10 in_*, 11..15 e0_*, 16..20 e1_*, 21..25 d0_*, 26..30 d1_*
    // 31..33 bn_Wl/bn_bl/bn_Wr, 34..36 out_*, 37..40 fW1/fb1/fW2/fb2

    // -------- workspace layout (floats) --------
    float* ws = (float*)d_ws;
    const long long S0 = (long long)N0C * 128;   // 10.24M floats
    const long long S1 = (long long)N1C * 256;
    const long long S2 = (long long)N2C * 256;
    float* W0 = ws;              float* W1 = ws + S0;
    float* W2 = ws + 2 * S0;     float* W3 = ws + 3 * S0;
    float* V0 = ws + 4 * S0;          float* V1 = V0 + S1;
    float* V2 = V0 + 2 * S1;          float* V3 = V0 + 3 * S1;
    float* U0 = ws + 4 * S0 + 4 * S1; float* U1 = U0 + S2; float* U2 = U0 + 2 * S2;
    float* haf = U0 + 3 * S2;                 // Ha: S0 halves = S0/2 floats
    _Float16* Ha = (_Float16*)haf;
    float* hbf = haf + S0 / 2;
    _Float16* Hb = (_Float16*)hbf;
    float* wpf = hbf + S0 / 2;                // wpack: up to 65536 halves
    _Float16* wpack = (_Float16*)wpf;
    float* cnt0  = wpf + 32768;
    float* cnt1  = cnt0 + 80000;
    float* cnt2  = cnt1 + 15040;
    float* tc1   = cnt2 + 3840;
    float* tc2   = tc1 + 15040;
    float* stats = tc2 + 3840;                // 512 floats

    // -------- degree / pooling counts --------
    k_zero<<<blocksFor(N0C, 256), 256, 0, stream>>>(cnt0, N0C);
    k_count<<<blocksFor(E0C, 256), 256, 0, stream>>>(edges0 + E0C, E0C, cnt0);
    k_zero<<<blocksFor(N1C, 256), 256, 0, stream>>>(cnt1, N1C);
    k_count<<<blocksFor(E1C, 256), 256, 0, stream>>>(edges1 + E1C, E1C, cnt1);
    k_zero<<<blocksFor(N2C, 256), 256, 0, stream>>>(cnt2, N2C);
    k_count<<<blocksFor(E2C, 256), 256, 0, stream>>>(edges2 + E2C, E2C, cnt2);
    k_zero<<<blocksFor(N1C, 256), 256, 0, stream>>>(tc1, N1C);
    k_count<<<blocksFor(N0C, 256), 256, 0, stream>>>(trace0, N0C, tc1);
    k_zero<<<blocksFor(N2C, 256), 256, 0, stream>>>(tc2, N2C);
    k_count<<<blocksFor(N1C, 256), 256, 0, stream>>>(trace1, N1C, tc2);

    // -------- input block: x(N0,3) -> W1(N0,64) (small-K path) --------
    run_block(x, edges0, E0C, cnt0, P[6], P[7], P[8], P[9], P[10],
              N0C, 3, NGF, /*agg*/W2, /*T*/W1, /*S*/W0, stats, /*out*/W1,
              Ha, Hb, wpack, stream);

    // -------- encoder e0: W1(64) -> W0(128), skip0 = W0 --------
    run_block(W1, edges0, E0C, cnt0, P[11], P[12], P[13], P[14], P[15],
              N0C, NGF, 2 * NGF, W2, W3, W0, stats, W0, Ha, Hb, wpack, stream);

    // pool level0 -> level1: V0(N1,128)
    seg_mean(W0, trace0, tc1, N0C, N1C, 2 * NGF, V0, stream);

    // -------- encoder e1: V0(128) -> V2(256), skip1 = V2 --------
    run_block(V0, edges1, E1C, cnt1, P[16], P[17], P[18], P[19], P[20],
              N1C, 2 * NGF, 4 * NGF, V1, V3, V2, stats, V2, Ha, Hb, wpack, stream);

    // pool level1 -> level2: U0(N2,256)
    seg_mean(V2, trace1, tc2, N1C, N2C, 4 * NGF, U0, stream);

    // -------- bottleneck: 6 residual blocks at 256 --------
    const float* bnWl = P[31];
    const float* bnbl = P[32];
    const float* bnWr = P[33];
    float* hcur = U0; float* hnext = U1;
    for (int i = 0; i < 6; ++i) {
        run_block(hcur, edges2, E2C, cnt2,
                  bnWl + (long long)i * 256 * 256, bnbl + (long long)i * 256,
                  bnWr + (long long)i * 256 * 256,
                  nullptr, nullptr,
                  N2C, 4 * NGF, 4 * NGF, U2, hnext, nullptr, stats, hnext,
                  Ha, Hb, wpack, stream);
        float* t = hcur; hcur = hnext; hnext = t;
    }
    // after 6 swaps hcur == U0

    // unpool -> level1, add skip1: V0 = hcur[trace1] + V2
    k_gather_add<<<N1C, 4 * NGF, 0, stream>>>(hcur, trace1, V2, V0);

    // -------- decoder d0: V0(256) -> V3(128) --------
    run_block(V0, edges1, E1C, cnt1, P[21], P[22], P[23], P[24], P[25],
              N1C, 4 * NGF, 2 * NGF, V1, V3, V2, stats, V3, Ha, Hb, wpack, stream);

    // unpool -> level0, add skip0: W1 = V3[trace0] + W0
    k_gather_add<<<N0C, 2 * NGF, 0, stream>>>(V3, trace0, W0, W1);

    // -------- decoder d1: W1(128) -> W3(64) --------
    run_block(W1, edges0, E0C, cnt0, P[26], P[27], P[28], P[29], P[30],
              N0C, 2 * NGF, NGF, W2, W3, W0, stats, W3, Ha, Hb, wpack, stream);

    // -------- output conv block (no shortcut): W3(64) -> W2(64) --------
    run_block(W3, edges0, E0C, cnt0, P[34], P[35], P[36], nullptr, nullptr,
              N0C, NGF, NGF, W1, W2, nullptr, stats, W2, Ha, Hb, wpack, stream);

    // -------- final head: elu(inorm(h @ fW1 + fb1)) -> tanh(@ fW2 + fb2) --------
    k_cvt_f16<<<blocksFor((long long)N0C * NGF, 256), 256, 0, stream>>>(
        W2, Ha, (long long)N0C * NGF);
    gemm16(Ha, P[37], P[38], W0, N0C, NGF, NGF, 0, wpack, stream);
    k_zero<<<1, 256, 0, stream>>>(stats, 2 * NGF);
    k_stats<<<blocksFor(N0C, 512), NGF, 0, stream>>>(W0, N0C, 512, stats);
    k_norm_elu_res<<<N0C, NGF, 0, stream>>>(W0, nullptr, stats, N0C, W0);

    k_cvt_f16<<<blocksFor((long long)N0C * NGF, 256), 256, 0, stream>>>(
        W0, Ha, (long long)N0C * NGF);
    float* outp = (float*)d_out;
    gemm16(Ha, P[39], P[40], outp, N0C, NGF, 3, 0, wpack, stream);
    k_tanh<<<blocksFor((long long)N0C * 3, 256), 256, 0, stream>>>(outp, (long long)N0C * 3);
}
